// TurnGPTAttention_50053548867726
// MI455X (gfx1250) — compile-verified, matrix operand order checked by reading, not run on hardware
//
#include <hip/hip_runtime.h>
#include <hip/hip_bf16.h>
#include <math.h>

#define BB 2
#define TT 2048
#define CC 1024
#define HH 16
#define DD 64
#define N3 (3 * CC)

typedef __attribute__((ext_vector_type(16))) __bf16 v16bf;
typedef __attribute__((ext_vector_type(8)))  float  v8f;

// ---------- helpers ----------

__device__ __forceinline__ unsigned short f32_to_bf16(float f) {
    unsigned int u = __float_as_uint(f);
    unsigned int r = 0x7FFFu + ((u >> 16) & 1u);   // round-to-nearest-even
    u += r;
    return (unsigned short)(u >> 16);
}

// Load a 16x32 bf16 A-fragment (or mirrored 32x16 B-fragment) for one lane.
// rowPtr = base of this lane's row (row index = lane & 15) in a row-major
// matrix. Per the CDNA5 ISA layout, the lane's 16 elements are two contiguous
// 16-byte chunks at k0 + half*8 and k0 + 16 + half*8 (half = lane>>4).
__device__ __forceinline__ v16bf load_frag(const unsigned short* rowPtr, int k0, int half) {
    union { v16bf v; uint4 q[2]; } f;
    f.q[0] = *reinterpret_cast<const uint4*>(rowPtr + k0 + half * 8);
    f.q[1] = *reinterpret_cast<const uint4*>(rowPtr + k0 + 16 + half * 8);
    return f.v;
}

__device__ __forceinline__ v8f wmma_bf16(v16bf a, v16bf b, v8f c) {
    return __builtin_amdgcn_wmma_f32_16x16x32_bf16(false, a, false, b, (short)0, c, false, false);
}

// Async copy 16B global -> LDS (ASYNCcnt-tracked, no VGPR round trip).
// lds_off = LDS byte address (low 32 bits of generic shared pointer).
__device__ __forceinline__ void async_load_b128(unsigned lds_off, const unsigned short* g) {
    asm volatile("global_load_async_to_lds_b128 %0, %1, off"
                 :: "v"(lds_off), "v"(g) : "memory");
}

__device__ __forceinline__ void wait_asynccnt0() {
    asm volatile("s_wait_asynccnt 0x0" ::: "memory");
}

// ---------- prep kernels ----------

__global__ void k_cvt_bf16(const float* __restrict__ in, unsigned short* __restrict__ out, int n) {
    int i = blockIdx.x * blockDim.x + threadIdx.x;
    if (i < n) out[i] = f32_to_bf16(in[i]);
}

// in: [K][N] f32 row-major  ->  out: [N][K] bf16 row-major (W transpose)
__global__ void k_transpose_bf16(const float* __restrict__ in, unsigned short* __restrict__ out,
                                 int K, int N) {
    int i = blockIdx.x * blockDim.x + threadIdx.x;
    if (i < N * K) {
        int n = i / K;
        int k = i - n * K;
        out[i] = f32_to_bf16(in[(size_t)k * N + n]);
    }
}

// ---------- QKV GEMM: qkv = x @ W_attn + b_attn, scattered to heads ----------
// One wave computes a 32x64 tile: 2 A-frags x 4 B-frags -> 8 WMMA per k-step.
__global__ __launch_bounds__(32) void k_qkv_gemm(const unsigned short* __restrict__ xb,
                                                 const unsigned short* __restrict__ WtA,
                                                 const float* __restrict__ b_attn,
                                                 unsigned short* __restrict__ Qb,
                                                 unsigned short* __restrict__ Kb,
                                                 unsigned short* __restrict__ Vt) {
    const int lane = threadIdx.x & 31;
    const int half = lane >> 4;
    const int nl   = lane & 15;
    const int NT   = N3 / 64;                // 48 column tiles of 64
    const int mt   = blockIdx.x / NT;
    const int nt   = blockIdx.x - mt * NT;
    const int m0   = mt * 32;
    const int n0   = nt * 64;

    v8f acc[2][4] = {};
    const unsigned short* arow0 = xb + (size_t)(m0 + nl) * CC;
    const unsigned short* arow1 = xb + (size_t)(m0 + 16 + nl) * CC;

    for (int k0 = 0; k0 < CC; k0 += 32) {
        v16bf a0 = load_frag(arow0, k0, half);
        v16bf a1 = load_frag(arow1, k0, half);
#pragma unroll
        for (int j = 0; j < 4; ++j) {
            v16bf b = load_frag(WtA + (size_t)(n0 + j * 16 + nl) * CC, k0, half);
            acc[0][j] = wmma_bf16(a0, b, acc[0][j]);
            acc[1][j] = wmma_bf16(a1, b, acc[1][j]);
        }
    }

#pragma unroll
    for (int mi = 0; mi < 2; ++mi) {
#pragma unroll
        for (int j = 0; j < 4; ++j) {
            const int n     = n0 + j * 16 + nl;
            const float bia = b_attn[n];
            const int which = n / CC;             // 0=q 1=k 2=v
            const int c     = n - which * CC;
            const int h     = c / DD;
            const int d     = c - h * DD;
#pragma unroll
            for (int r = 0; r < 8; ++r) {
                const int m  = m0 + mi * 16 + r + 8 * half;
                const int bI = m / TT;
                const int t  = m - bI * TT;
                const unsigned short v = f32_to_bf16(acc[mi][j][r] + bia);
                if (which == 0)      Qb[(((size_t)bI * HH + h) * TT + t) * DD + d] = v;
                else if (which == 1) Kb[(((size_t)bI * HH + h) * TT + t) * DD + d] = v;
                else                 Vt[(((size_t)bI * HH + h) * DD + d) * TT + t] = v;
            }
        }
    }
}

// ---------- Flash attention ----------
// 4-wave workgroup covers 64 query rows of one (b,h). K/V 32-key blocks are
// staged cooperatively into LDS with async global->LDS copies and shared by
// all 4 waves; each wave then runs 16 query rows of online-softmax WMMA.
__global__ __launch_bounds__(128) void k_flash(const unsigned short* __restrict__ Qb,
                                               const unsigned short* __restrict__ Kb,
                                               const unsigned short* __restrict__ Vt,
                                               unsigned short* __restrict__ Yh) {
    const int tid  = threadIdx.x;
    const int lane = tid & 31;
    const int w    = tid >> 5;               // wave id 0..3
    const int half = lane >> 4;
    const int nl   = lane & 15;
    const int QB   = TT / 64;
    const int qb   = blockIdx.x % QB;
    const int bh   = blockIdx.x / QB;
    const int b    = bh / HH;
    const int h    = bh - b * HH;
    const int q0b  = qb * 64;                // block's first query row
    const int q0   = q0b + w * 16;           // this wave's first query row

    const unsigned short* Qh = Qb + (size_t)bh * TT * DD;
    const unsigned short* Kh = Kb + (size_t)bh * TT * DD;
    const unsigned short* Vh = Vt + (size_t)bh * DD * TT;

    // LDS: K tile [32 keys][64 d] + V tile [64 d][32 keys] + per-wave P bounce
    __shared__ __align__(16) unsigned short smem[2048 + 2048 + 4 * 512];
    unsigned short* Kl = smem;
    unsigned short* Vl = smem + 2048;
    unsigned short* Pl = smem + 4096 + w * 512;

    const unsigned short* qrow = Qh + (size_t)(q0 + nl) * DD;
    const v16bf aq0 = load_frag(qrow, 0, half);
    const v16bf aq1 = load_frag(qrow, 32, half);

    float mrow[8], lrow[8];
#pragma unroll
    for (int r = 0; r < 8; ++r) { mrow[r] = -__builtin_inff(); lrow[r] = 0.f; }
    v8f O[4] = {};

    const float scale = 0.125f;               // 1/sqrt(64)

    for (int kb = 0; kb <= q0b + 63; kb += 32) {
        // ---- cooperative async stage of K (32x64) and V (64x32) into LDS ----
#pragma unroll
        for (int i = tid; i < 256; i += 128) {     // K tile: 256 x 16B
            const int row = i >> 3, ch = i & 7;
            const unsigned loff = (unsigned)(uintptr_t)(Kl + row * 64 + ch * 8);
            async_load_b128(loff, Kh + (size_t)(kb + row) * DD + ch * 8);
        }
#pragma unroll
        for (int i = tid; i < 256; i += 128) {     // V tile: 256 x 16B
            const int d = i >> 2, ch = i & 3;
            const unsigned loff = (unsigned)(uintptr_t)(Vl + d * 32 + ch * 8);
            async_load_b128(loff, Vh + (size_t)d * TT + kb + ch * 8);
        }
        wait_asynccnt0();
        __syncthreads();

        if (kb <= q0 + 15) {                  // causal: this wave needs this block
            // ---- S = scale * Q K^T  (16q x 32k), causal-masked ----
            v8f s[2];
#pragma unroll
            for (int nt = 0; nt < 2; ++nt) {
                const unsigned short* krow = Kl + (size_t)(nt * 16 + nl) * 64;
                v16bf b0 = load_frag(krow, 0, half);
                v16bf b1 = load_frag(krow, 32, half);
                v8f c = {};
                c = wmma_bf16(aq0, b0, c);
                c = wmma_bf16(aq1, b1, c);
                s[nt] = c;
            }
#pragma unroll
            for (int nt = 0; nt < 2; ++nt) {
                const int k = kb + nt * 16 + nl;
#pragma unroll
                for (int r = 0; r < 8; ++r) {
                    const int q = q0 + r + 8 * half;
                    float sv = s[nt][r] * scale;
                    if (k > q) sv = -__builtin_inff();
                    s[nt][r] = sv;
                }
            }
            // ---- online softmax (row stats across the 16 lanes of this half) ----
            float rmax[8];
#pragma unroll
            for (int r = 0; r < 8; ++r) rmax[r] = fmaxf(s[0][r], s[1][r]);
#pragma unroll
            for (int msk = 1; msk < 16; msk <<= 1)
#pragma unroll
                for (int r = 0; r < 8; ++r) rmax[r] = fmaxf(rmax[r], __shfl_xor(rmax[r], msk, 32));

            float alpha[8];
#pragma unroll
            for (int r = 0; r < 8; ++r) {
                const float nm = fmaxf(mrow[r], rmax[r]);
                alpha[r] = __expf(mrow[r] - nm);
                mrow[r] = nm;
            }
#pragma unroll
            for (int nt = 0; nt < 2; ++nt)
#pragma unroll
                for (int r = 0; r < 8; ++r) s[nt][r] = __expf(s[nt][r] - mrow[r]);

            float psum[8];
#pragma unroll
            for (int r = 0; r < 8; ++r) psum[r] = s[0][r] + s[1][r];
#pragma unroll
            for (int msk = 1; msk < 16; msk <<= 1)
#pragma unroll
                for (int r = 0; r < 8; ++r) psum[r] += __shfl_xor(psum[r], msk, 32);
#pragma unroll
            for (int r = 0; r < 8; ++r) lrow[r] = lrow[r] * alpha[r] + psum[r];
#pragma unroll
            for (int j = 0; j < 4; ++j)
#pragma unroll
                for (int r = 0; r < 8; ++r) O[j][r] *= alpha[r];

            // ---- P: D-layout -> A-layout via wave-private LDS bounce ----
            // (same-wave LDS ops are in-order; no workgroup barrier needed)
#pragma unroll
            for (int nt = 0; nt < 2; ++nt)
#pragma unroll
                for (int r = 0; r < 8; ++r)
                    Pl[(r + 8 * half) * 32 + nt * 16 + nl] = f32_to_bf16(s[nt][r]);
            const v16bf pf = load_frag(Pl + nl * 32, 0, half);

            // ---- O += P V  (V tile is [d][key] in LDS) ----
#pragma unroll
            for (int j = 0; j < 4; ++j) {
                v16bf vf = load_frag(Vl + (size_t)(j * 16 + nl) * 32, 0, half);
                O[j] = wmma_bf16(pf, vf, O[j]);
            }
        }
        __syncthreads();                      // protect LDS tiles before next stage
    }

    // ---- normalize, write y_heads [B*T][C] bf16 ----
    float inv[8];
#pragma unroll
    for (int r = 0; r < 8; ++r) inv[r] = 1.0f / lrow[r];
#pragma unroll
    for (int j = 0; j < 4; ++j) {
        const int col = h * DD + j * 16 + nl;
#pragma unroll
        for (int r = 0; r < 8; ++r) {
            const int rowY = b * TT + q0 + r + 8 * half;
            Yh[(size_t)rowY * CC + col] = f32_to_bf16(O[j][r] * inv[r]);
        }
    }
}

// ---------- Projection GEMM: out = y_heads @ W_proj + b_proj (fp32 out) ----------
__global__ __launch_bounds__(32) void k_proj_gemm(const unsigned short* __restrict__ Yh,
                                                  const unsigned short* __restrict__ WtP,
                                                  const float* __restrict__ b_proj,
                                                  float* __restrict__ out) {
    const int lane = threadIdx.x & 31;
    const int half = lane >> 4;
    const int nl   = lane & 15;
    const int NT   = CC / 64;                 // 16 column tiles
    const int mt   = blockIdx.x / NT;
    const int nt   = blockIdx.x - mt * NT;
    const int m0   = mt * 32;
    const int n0   = nt * 64;

    v8f acc[2][4] = {};
    const unsigned short* arow0 = Yh + (size_t)(m0 + nl) * CC;
    const unsigned short* arow1 = Yh + (size_t)(m0 + 16 + nl) * CC;

    for (int k0 = 0; k0 < CC; k0 += 32) {
        v16bf a0 = load_frag(arow0, k0, half);
        v16bf a1 = load_frag(arow1, k0, half);
#pragma unroll
        for (int j = 0; j < 4; ++j) {
            v16bf b = load_frag(WtP + (size_t)(n0 + j * 16 + nl) * CC, k0, half);
            acc[0][j] = wmma_bf16(a0, b, acc[0][j]);
            acc[1][j] = wmma_bf16(a1, b, acc[1][j]);
        }
    }

#pragma unroll
    for (int mi = 0; mi < 2; ++mi) {
#pragma unroll
        for (int j = 0; j < 4; ++j) {
            const int n = n0 + j * 16 + nl;
            const float bia = b_proj[n];
#pragma unroll
            for (int r = 0; r < 8; ++r) {
                const int m = m0 + mi * 16 + r + 8 * half;
                out[(size_t)m * CC + n] = acc[mi][j][r] + bia;
            }
        }
    }
}

// ---------- host launcher ----------

extern "C" void kernel_launch(void* const* d_in, const int* in_sizes, int n_in,
                              void* d_out, int out_size, void* d_ws, size_t ws_size,
                              hipStream_t stream) {
    const float* x      = (const float*)d_in[0];
    const float* W_attn = (const float*)d_in[1];
    const float* b_attn = (const float*)d_in[2];
    const float* W_proj = (const float*)d_in[3];
    const float* b_proj = (const float*)d_in[4];
    float* out = (float*)d_out;

    const size_t M = (size_t)BB * TT;            // 4096

    char* ws = (char*)d_ws;
    size_t off = 0;
    auto alloc = [&](size_t bytes) { char* p = ws + off; off = (off + bytes + 255) & ~(size_t)255; return p; };

    unsigned short* xb  = (unsigned short*)alloc(M * CC * 2);          // 8 MiB
    unsigned short* WtA = (unsigned short*)alloc((size_t)N3 * CC * 2); // 6 MiB
    unsigned short* WtP = (unsigned short*)alloc((size_t)CC * CC * 2); // 2 MiB
    unsigned short* Qb  = (unsigned short*)alloc((size_t)BB * HH * TT * DD * 2);
    unsigned short* Kb  = (unsigned short*)alloc((size_t)BB * HH * TT * DD * 2);
    unsigned short* Vt  = (unsigned short*)alloc((size_t)BB * HH * DD * TT * 2);
    unsigned short* Yh  = (unsigned short*)alloc(M * CC * 2);          // 8 MiB
    (void)ws_size; (void)in_sizes; (void)n_in; (void)out_size;

    // prep: convert x, transpose+convert weights
    {
        int n = (int)(M * CC);
        k_cvt_bf16<<<(n + 255) / 256, 256, 0, stream>>>(x, xb, n);
    }
    {
        int n = N3 * CC;
        k_transpose_bf16<<<(n + 255) / 256, 256, 0, stream>>>(W_attn, WtA, CC, N3);
    }
    {
        int n = CC * CC;
        k_transpose_bf16<<<(n + 255) / 256, 256, 0, stream>>>(W_proj, WtP, CC, CC);
    }

    // QKV GEMM: (M/32) x (3C/64) waves
    k_qkv_gemm<<<dim3((M / 32) * (N3 / 64)), 32, 0, stream>>>(xb, WtA, b_attn, Qb, Kb, Vt);

    // flash attention: B*H*(T/64) blocks of 4 waves
    k_flash<<<dim3(BB * HH * (TT / 64)), 128, 0, stream>>>(Qb, Kb, Vt, Yh);

    // projection: (M/32) x (C/64) waves
    k_proj_gemm<<<dim3((M / 32) * (CC / 64)), 32, 0, stream>>>(Yh, WtP, b_proj, out);
}
